// DinoDetrDecoderLayer_16080357556426
// MI455X (gfx1250) — matrix operand order repaired and checked by reading, not compiled, not run on hardware
//
#include <hip/hip_runtime.h>
#include <hip/hip_bf16.h>

#define NQ     900
#define BATCH  8
#define DMODEL 256
#define NH     8
#define DH     32
#define NLVL   4
#define NPT    4
#define DFFN   2048
#define STOT   12240
#define NROWS  (NQ * BATCH)     // 7200
#define VROWS  (STOT * BATCH)   // 97920

typedef __bf16 bf16_t;
typedef bf16_t v16bf __attribute__((ext_vector_type(16)));
typedef bf16_t v4bf  __attribute__((ext_vector_type(4)));
typedef float  v8f   __attribute__((ext_vector_type(8)));

__device__ __forceinline__ v8f wmma_bf16(v16bf a, v16bf b, v8f c) {
  return __builtin_amdgcn_wmma_f32_16x16x32_bf16(false, a, false, b, (short)0, c, false, false);
}

// A-matrix (16x32, 16-bit) K map: lanes 0-15 hold K {0..7, 16..23}, lanes 16-31 hold K {8..15, 24..31}
__device__ __forceinline__ int kmapA(int i, bool hi) {
  int k = (i < 4) ? (2 * i) : (16 + 2 * (i - 4));
  return hi ? (k + 8) : k;
}
// B-matrix (32x16, 16-bit) K map: lanes 0-15 hold K {0..7, 16..23}, lanes 16-31 hold K {8..15, 24..31}
__device__ __forceinline__ int kmapB(int i, bool hi) {
  int k = (i < 4) ? (2 * i) : (16 + 2 * (i - 4));
  return hi ? (k + 8) : k;
}

__device__ __forceinline__ v4bf cvt4(float4 v) {
  v4bf r = { (bf16_t)v.x, (bf16_t)v.y, (bf16_t)v.z, (bf16_t)v.w };
  return r;
}

// ---------------------------------------------------------------------------
// Generic GEMM: C[M,N] = A[M,K] @ W[N,K]^T + bias, optional ReLU.
// f32 in memory, bf16 WMMA compute, f32 accumulate.
// Block: 256 threads (8 waves, 2x4). Block tile 64x128, K-step 32.
// Software pipeline: global->regs for tile t+1 overlaps WMMA on tile t,
// double-buffered LDS, 1 barrier per K-step. Out-of-range A rows are clamped
// (they only feed store-masked C rows), so tile loads are branch-free.
// ---------------------------------------------------------------------------
#define BM 64
#define BN 128
#define BK 32
#define LDP (BK + 8)   // padded row: 40 bf16 = 80B, keeps 16B alignment

__global__ __launch_bounds__(256) void gemm_bf16_wmma(
    const float* __restrict__ A, const float* __restrict__ W,
    const float* __restrict__ bias, float* __restrict__ C,
    int M, int N, int K, int do_relu) {
  __shared__ bf16_t As[2][BM][LDP];
  __shared__ bf16_t Bs[2][BN][LDP];

  const int tid  = threadIdx.x;
  const int wave = tid >> 5;
  const int lane = tid & 31;
  const bool hi  = lane >= 16;
  const int m0   = blockIdx.x * BM;
  const int n0   = blockIdx.y * BN;
  const int wrow = (wave >> 2) * 32;   // 0 or 32
  const int wcol = (wave & 3) * 32;    // 0,32,64,96

  // per-thread tile-load coordinates (float4 granularity, K contiguous)
  const int lrA = tid >> 3;            // row 0..31 (second half at +32)
  const int lcA = (tid & 7) * 4;       // col 0..28
  int agr0 = m0 + lrA;      if (agr0 >= M) agr0 = M - 1;
  int agr1 = m0 + lrA + 32; if (agr1 >= M) agr1 = M - 1;
  const float* aP0 = A + (size_t)agr0 * K + lcA;
  const float* aP1 = A + (size_t)agr1 * K + lcA;
  const float* wP0 = W + (size_t)(n0 + lrA +  0) * K + lcA;
  const float* wP1 = W + (size_t)(n0 + lrA + 32) * K + lcA;
  const float* wP2 = W + (size_t)(n0 + lrA + 64) * K + lcA;
  const float* wP3 = W + (size_t)(n0 + lrA + 96) * K + lcA;

  float4 ar0, ar1, wr0, wr1, wr2, wr3;
  auto gload = [&](int kt) {
    const size_t ko = (size_t)kt * BK;
    ar0 = *(const float4*)(aP0 + ko);
    ar1 = *(const float4*)(aP1 + ko);
    wr0 = *(const float4*)(wP0 + ko);
    wr1 = *(const float4*)(wP1 + ko);
    wr2 = *(const float4*)(wP2 + ko);
    wr3 = *(const float4*)(wP3 + ko);
  };
  auto sstore = [&](int buf) {
    *(v4bf*)&As[buf][lrA][lcA]        = cvt4(ar0);
    *(v4bf*)&As[buf][lrA + 32][lcA]   = cvt4(ar1);
    *(v4bf*)&Bs[buf][lrA +  0][lcA]   = cvt4(wr0);
    *(v4bf*)&Bs[buf][lrA + 32][lcA]   = cvt4(wr1);
    *(v4bf*)&Bs[buf][lrA + 64][lcA]   = cvt4(wr2);
    *(v4bf*)&Bs[buf][lrA + 96][lcA]   = cvt4(wr3);
  };

  v8f acc00 = {0.f,0.f,0.f,0.f,0.f,0.f,0.f,0.f};
  v8f acc01 = acc00, acc10 = acc00, acc11 = acc00;
  const int ar = lane & 15;

  const int ntiles = K / BK;
  gload(0);
  sstore(0);
  for (int t = 0; t < ntiles; ++t) {
    __syncthreads();
    const int cur = t & 1;
    if (t + 1 < ntiles) gload(t + 1);   // in flight during compute

    v16bf af0, af1, bf0, bf1;
#pragma unroll
    for (int i = 0; i < 8; ++i) {
      int ka = kmapA(i, hi);
      int kb = kmapB(i, hi);
      af0[2 * i]     = As[cur][wrow + ar][ka];
      af0[2 * i + 1] = As[cur][wrow + ar][ka + 1];
      af1[2 * i]     = As[cur][wrow + 16 + ar][ka];
      af1[2 * i + 1] = As[cur][wrow + 16 + ar][ka + 1];
      bf0[2 * i]     = Bs[cur][wcol + ar][kb];
      bf0[2 * i + 1] = Bs[cur][wcol + ar][kb + 1];
      bf1[2 * i]     = Bs[cur][wcol + 16 + ar][kb];
      bf1[2 * i + 1] = Bs[cur][wcol + 16 + ar][kb + 1];
    }
    acc00 = wmma_bf16(af0, bf0, acc00);
    acc01 = wmma_bf16(af0, bf1, acc01);
    acc10 = wmma_bf16(af1, bf0, acc10);
    acc11 = wmma_bf16(af1, bf1, acc11);

    if (t + 1 < ntiles) sstore(cur ^ 1);
  }

  // C/D layout: lanes 0-15: vgpr j -> row j, col lane; lanes 16-31: row j+8, col lane-16
  const int c0 = n0 + wcol + (lane & 15);
  const int c1 = c0 + 16;
  const float bv0 = bias ? bias[c0] : 0.f;
  const float bv1 = bias ? bias[c1] : 0.f;
  const int rbase = m0 + wrow + (hi ? 8 : 0);
#pragma unroll
  for (int j = 0; j < 8; ++j) {
    float v00 = acc00[j] + bv0, v01 = acc01[j] + bv1;
    float v10 = acc10[j] + bv0, v11 = acc11[j] + bv1;
    if (do_relu) {
      v00 = fmaxf(v00, 0.f); v01 = fmaxf(v01, 0.f);
      v10 = fmaxf(v10, 0.f); v11 = fmaxf(v11, 0.f);
    }
    int r0 = rbase + j, r1 = rbase + 16 + j;
    if (r0 < M) { C[(size_t)r0 * N + c0] = v00; C[(size_t)r0 * N + c1] = v01; }
    if (r1 < M) { C[(size_t)r1 * N + c0] = v10; C[(size_t)r1 * N + c1] = v11; }
  }
}

// ---------------------------------------------------------------------------
// Flash attention over the packed QKV buffer ([NQ*B rows][768]).
// One wave per (16-query tile, b, h). Key tiles of 32:
//   2 score WMMAs (16x16x32) -> online softmax -> LDS relayout -> 2 PV WMMAs.
// ---------------------------------------------------------------------------
__global__ __launch_bounds__(32) void attn_flash(
    const float* __restrict__ qkv, float* __restrict__ out) {
  const int qt = blockIdx.x;          // query tile (16 queries)
  const int bh = blockIdx.y;
  const int b  = bh / NH, h = bh % NH;
  const int lane = threadIdx.x;
  const bool hi  = lane >= 16;
  const int lrow = lane & 15;

  const float scale = 0.17677669529663687f; // 1/sqrt(32)

  // Q fragment (A operand, 16x32), scale folded in
  v16bf qa;
  {
    int qrow = qt * 16 + lrow;
    if (qrow > NQ - 1) qrow = NQ - 1;
    const float* qb = qkv + ((size_t)qrow * BATCH + b) * 768 + h * DH;
#pragma unroll
    for (int i = 0; i < 8; ++i) {
      int k = kmapA(i, hi);
      qa[2 * i]     = (bf16_t)(qb[k] * scale);
      qa[2 * i + 1] = (bf16_t)(qb[k + 1] * scale);
    }
  }

  v8f o0 = {0.f,0.f,0.f,0.f,0.f,0.f,0.f,0.f};
  v8f o1 = o0;
  float m_[8], l_[8];
#pragma unroll
  for (int j = 0; j < 8; ++j) { m_[j] = -3.0e38f; l_[j] = 0.f; }

  __shared__ bf16_t Pl[16][32];

  const int ntiles = (NQ + 31) / 32; // 29
  for (int kt = 0; kt < ntiles; ++kt) {
    const int key0 = kt * 32 + lrow;
    const int key1 = key0 + 16;
    const int kc0 = key0 < NQ ? key0 : NQ - 1;
    const int kc1 = key1 < NQ ? key1 : NQ - 1;
    const float* kb0 = qkv + ((size_t)kc0 * BATCH + b) * 768 + 256 + h * DH;
    const float* kb1 = qkv + ((size_t)kc1 * BATCH + b) * 768 + 256 + h * DH;
    v16bf kf0, kf1;
#pragma unroll
    for (int i = 0; i < 8; ++i) {
      int k = kmapB(i, hi);
      kf0[2 * i]     = (bf16_t)kb0[k];
      kf0[2 * i + 1] = (bf16_t)kb0[k + 1];
      kf1[2 * i]     = (bf16_t)kb1[k];
      kf1[2 * i + 1] = (bf16_t)kb1[k + 1];
    }
    v8f zero = {0.f,0.f,0.f,0.f,0.f,0.f,0.f,0.f};
    v8f s0 = wmma_bf16(qa, kf0, zero);
    v8f s1 = wmma_bf16(qa, kf1, zero);

    const bool bad0 = key0 >= NQ, bad1 = key1 >= NQ;
#pragma unroll
    for (int j = 0; j < 8; ++j) {
      if (bad0) s0[j] = -1.0e30f;
      if (bad1) s1[j] = -1.0e30f;
    }

    // online softmax, per row (row = j + 8*hi, shared by the 16 lanes of a half)
#pragma unroll
    for (int j = 0; j < 8; ++j) {
      float mx = fmaxf(s0[j], s1[j]);
      for (int d = 1; d < 16; d <<= 1) mx = fmaxf(mx, __shfl_xor(mx, d));
      float mn = fmaxf(m_[j], mx);
      float alpha = __expf(m_[j] - mn);
      float p0 = __expf(s0[j] - mn);
      float p1 = __expf(s1[j] - mn);
      float rs = p0 + p1;
      for (int d = 1; d < 16; d <<= 1) rs += __shfl_xor(rs, d);
      l_[j] = l_[j] * alpha + rs;
      m_[j] = mn;
      o0[j] *= alpha;
      o1[j] *= alpha;
      int pr = j + (hi ? 8 : 0);
      Pl[pr][lrow]      = (bf16_t)p0;
      Pl[pr][lrow + 16] = (bf16_t)p1;
    }
    __syncthreads();

    // P as A operand (16 queries x 32 keys)
    v16bf pf;
#pragma unroll
    for (int i = 0; i < 8; ++i) {
      int k = kmapA(i, hi);
      pf[2 * i]     = Pl[lrow][k];
      pf[2 * i + 1] = Pl[lrow][k + 1];
    }
    // V fragments (B operand, key x d): lane = d column
    v16bf vf0, vf1;
#pragma unroll
    for (int i = 0; i < 8; ++i) {
      int kk = kmapB(i, hi);
      int ky0 = kt * 32 + kk;     if (ky0 > NQ - 1) ky0 = NQ - 1;
      int ky1 = kt * 32 + kk + 1; if (ky1 > NQ - 1) ky1 = NQ - 1;
      const float* v0p = qkv + ((size_t)ky0 * BATCH + b) * 768 + 512 + h * DH;
      const float* v1p = qkv + ((size_t)ky1 * BATCH + b) * 768 + 512 + h * DH;
      vf0[2 * i]     = (bf16_t)v0p[lrow];
      vf0[2 * i + 1] = (bf16_t)v1p[lrow];
      vf1[2 * i]     = (bf16_t)v0p[lrow + 16];
      vf1[2 * i + 1] = (bf16_t)v1p[lrow + 16];
    }
    o0 = wmma_bf16(pf, vf0, o0);
    o1 = wmma_bf16(pf, vf1, o1);
    __syncthreads();
  }

  // finalize: divide by l, write out[(q*B+b)*256 + h*32 + d]
#pragma unroll
  for (int j = 0; j < 8; ++j) {
    int qg = qt * 16 + j + (hi ? 8 : 0);
    if (qg < NQ) {
      float invl = 1.f / l_[j];
      float* op = out + ((size_t)qg * BATCH + b) * DMODEL + h * DH;
      op[lrow]      = o0[j] * invl;
      op[lrow + 16] = o1[j] * invl;
    }
  }
}

// ---------------------------------------------------------------------------
// Deformable-attention bilinear sampling. One wave per (q*B+b, h); lane = d.
// ---------------------------------------------------------------------------
__global__ __launch_bounds__(256) void deform_sample(
    const float* __restrict__ value,  // [S*B][256], row = s*B+b
    const float* __restrict__ offs,   // [NQ*B][256]
    const float* __restrict__ awl,    // [NQ*B][128] raw logits
    const float* __restrict__ ref,    // [NQ*B][4]
    float* __restrict__ out) {        // [NQ*B][256]
  const int gw = blockIdx.x * 8 + (threadIdx.x >> 5);
  if (gw >= NROWS * NH) return;
  const int lane = threadIdx.x & 31;       // d
  const int h  = gw % NH;
  const int qb = gw / NH;                  // q*B + b
  const int b  = qb % BATCH;

  const float rx = ref[(size_t)qb * 4 + 0];
  const float ry = ref[(size_t)qb * 4 + 1];
  const float rw = ref[(size_t)qb * 4 + 2];
  const float rh = ref[(size_t)qb * 4 + 3];

  // softmax over the 16 attention logits of this head (redundant per lane, cheap)
  const float* awp = awl + (size_t)qb * 128 + h * 16;
  float wmax = -3.0e38f;
#pragma unroll
  for (int t = 0; t < 16; ++t) wmax = fmaxf(wmax, awp[t]);
  float wv[16], wsum = 0.f;
#pragma unroll
  for (int t = 0; t < 16; ++t) { wv[t] = __expf(awp[t] - wmax); wsum += wv[t]; }
  const float winv = 1.f / wsum;

  const int lvl_start[NLVL] = {0, 9216, 11520, 12096};
  const int lvl_hw[NLVL]    = {96, 48, 24, 12};
  const float* op = offs + (size_t)qb * 256 + h * (NLVL * NPT * 2);

  float acc = 0.f;
  for (int l = 0; l < NLVL; ++l) {
    const int hw = lvl_hw[l], base = lvl_start[l];
    for (int p = 0; p < NPT; ++p) {
      float ox = op[(l * NPT + p) * 2 + 0];
      float oy = op[(l * NPT + p) * 2 + 1];
      float lx = rx + ox * 0.125f * rw;    // offs/P * wh * 0.5
      float ly = ry + oy * 0.125f * rh;
      float x = lx * hw - 0.5f, y = ly * hw - 0.5f;
      float x0f = floorf(x), y0f = floorf(y);
      float fx = x - x0f, fy = y - y0f;
      int x0 = (int)x0f, y0 = (int)y0f;
      float wgt = wv[l * NPT + p] * winv;

      float s = 0.f;
#pragma unroll
      for (int cy = 0; cy < 2; ++cy) {
#pragma unroll
        for (int cx = 0; cx < 2; ++cx) {
          int xi = x0 + cx, yi = y0 + cy;
          bool ok = (xi >= 0) && (xi < hw) && (yi >= 0) && (yi < hw);
          if (ok) {
            float cw = (cx ? fx : 1.f - fx) * (cy ? fy : 1.f - fy);
            size_t row = (size_t)(base + yi * hw + xi) * BATCH + b;
            s += cw * value[row * DMODEL + h * DH + lane];
          }
        }
      }
      acc += wgt * s;
    }
  }
  out[(size_t)qb * DMODEL + h * DH + lane] = acc;
}

// ---------------------------------------------------------------------------
// Fused residual add + LayerNorm over D=256. One wave per row (8 rows/block).
// Each lane owns 8 contiguous columns (2 x float4).
// ---------------------------------------------------------------------------
__global__ __launch_bounds__(256) void add_layernorm(
    const float* __restrict__ x, const float* __restrict__ y,
    const float* __restrict__ g, const float* __restrict__ bta,
    float* __restrict__ out, int rows) {
  const int row = blockIdx.x * 8 + (threadIdx.x >> 5);
  if (row >= rows) return;
  const int lane = threadIdx.x & 31;
  const int cb = lane * 8;
  const float* xr = x + (size_t)row * DMODEL + cb;
  const float* yr = y + (size_t)row * DMODEL + cb;
  float4 a0 = *(const float4*)(xr + 0);
  float4 a1 = *(const float4*)(xr + 4);
  float4 b0 = *(const float4*)(yr + 0);
  float4 b1 = *(const float4*)(yr + 4);
  float v[8] = {a0.x + b0.x, a0.y + b0.y, a0.z + b0.z, a0.w + b0.w,
                a1.x + b1.x, a1.y + b1.y, a1.z + b1.z, a1.w + b1.w};
  float s = 0.f;
#pragma unroll
  for (int i = 0; i < 8; ++i) s += v[i];
  for (int d = 1; d < 32; d <<= 1) s += __shfl_xor(s, d);
  const float mean = s * (1.f / DMODEL);
  float vs = 0.f;
#pragma unroll
  for (int i = 0; i < 8; ++i) { float dd = v[i] - mean; vs += dd * dd; }
  for (int d = 1; d < 32; d <<= 1) vs += __shfl_xor(vs, d);
  const float rstd = rsqrtf(vs * (1.f / DMODEL) + 1e-5f);
  float4 g0 = *(const float4*)(g + cb);
  float4 g1 = *(const float4*)(g + cb + 4);
  float4 t0 = *(const float4*)(bta + cb);
  float4 t1 = *(const float4*)(bta + cb + 4);
  float4 o0 = make_float4((v[0]-mean)*rstd*g0.x+t0.x, (v[1]-mean)*rstd*g0.y+t0.y,
                          (v[2]-mean)*rstd*g0.z+t0.z, (v[3]-mean)*rstd*g0.w+t0.w);
  float4 o1 = make_float4((v[4]-mean)*rstd*g1.x+t1.x, (v[5]-mean)*rstd*g1.y+t1.y,
                          (v[6]-mean)*rstd*g1.z+t1.z, (v[7]-mean)*rstd*g1.w+t1.w);
  float* orow = out + (size_t)row * DMODEL + cb;
  *(float4*)(orow + 0) = o0;
  *(float4*)(orow + 4) = o1;
}

__global__ __launch_bounds__(256) void ew_add4(
    const float* __restrict__ a, const float* __restrict__ b,
    float* __restrict__ c, int n4) {
  int i = blockIdx.x * 256 + threadIdx.x;
  if (i < n4) {
    float4 x = ((const float4*)a)[i];
    float4 y = ((const float4*)b)[i];
    ((float4*)c)[i] = make_float4(x.x + y.x, x.y + y.y, x.z + y.z, x.w + y.w);
  }
}

// ---------------------------------------------------------------------------
extern "C" void kernel_launch(void* const* d_in, const int* in_sizes, int n_in,
                              void* d_out, int out_size, void* d_ws, size_t ws_size,
                              hipStream_t stream) {
  const float* queries = (const float*)d_in[0];
  const float* qpe     = (const float*)d_in[1];
  const float* ref     = (const float*)d_in[2];
  const float* memory  = (const float*)d_in[3];
  const float* in_w    = (const float*)d_in[7];
  const float* in_b    = (const float*)d_in[8];
  const float* out_w   = (const float*)d_in[9];
  const float* out_b   = (const float*)d_in[10];
  const float* n1g = (const float*)d_in[11], *n1b = (const float*)d_in[12];
  const float* n2g = (const float*)d_in[13], *n2b = (const float*)d_in[14];
  const float* n3g = (const float*)d_in[15], *n3b = (const float*)d_in[16];
  const float* l1w = (const float*)d_in[17], *l1b = (const float*)d_in[18];
  const float* l2w = (const float*)d_in[19], *l2b = (const float*)d_in[20];
  const float* ofw = (const float*)d_in[21], *ofb = (const float*)d_in[22];
  const float* aww = (const float*)d_in[23], *awb_ = (const float*)d_in[24];
  const float* vw  = (const float*)d_in[25], *vb  = (const float*)d_in[26];
  const float* cow = (const float*)d_in[27], *cob = (const float*)d_in[28];
  float* outp = (float*)d_out;

  float* f = (float*)d_ws;
  float* qk    = f; f += (size_t)NROWS * DMODEL;
  float* qkv   = f; f += (size_t)NROWS * 3 * DMODEL;
  float* sa    = f; f += (size_t)NROWS * DMODEL;
  float* sproj = f; f += (size_t)NROWS * DMODEL;
  float* q2    = f; f += (size_t)NROWS * DMODEL;
  float* qin   = f; f += (size_t)NROWS * DMODEL;
  float* val   = f; f += (size_t)VROWS * DMODEL;
  float* offb  = f; f += (size_t)NROWS * DMODEL;
  float* awlg  = f; f += (size_t)NROWS * 128;
  float* ca    = f; f += (size_t)NROWS * DMODEL;
  float* cproj = f; f += (size_t)NROWS * DMODEL;
  float* q3    = f; f += (size_t)NROWS * DMODEL;
  float* ffh   = f; f += (size_t)NROWS * DFFN;
  float* ffo   = f; f += (size_t)NROWS * DMODEL;

  const int n4 = (NROWS * DMODEL) / 4;
  dim3 blk(256);

  // 1) qk = queries + qpe
  ew_add4<<<dim3(n4 / 256), blk, 0, stream>>>(queries, qpe, qk, n4);
  // 2) qkv = qk @ in_w^T + in_b
  gemm_bf16_wmma<<<dim3((NROWS + BM - 1) / BM, (3 * DMODEL) / BN), blk, 0, stream>>>(
      qk, in_w, in_b, qkv, NROWS, 3 * DMODEL, DMODEL, 0);
  // 3) flash attention
  attn_flash<<<dim3((NQ + 15) / 16, BATCH * NH), dim3(32), 0, stream>>>(qkv, sa);
  // 4) sproj = sa @ out_w^T + out_b
  gemm_bf16_wmma<<<dim3((NROWS + BM - 1) / BM, DMODEL / BN), blk, 0, stream>>>(
      sa, out_w, out_b, sproj, NROWS, DMODEL, DMODEL, 0);
  // 5) q2 = LN(queries + sproj) [norm2]
  add_layernorm<<<dim3(NROWS / 8), blk, 0, stream>>>(queries, sproj, n2g, n2b, q2, NROWS);
  // 6) qin = q2 + qpe
  ew_add4<<<dim3(n4 / 256), blk, 0, stream>>>(q2, qpe, qin, n4);
  // 7) value projection: val = memory @ vw^T + vb  (bandwidth-critical)
  gemm_bf16_wmma<<<dim3(VROWS / BM, DMODEL / BN), blk, 0, stream>>>(
      memory, vw, vb, val, VROWS, DMODEL, DMODEL, 0);
  // 8) offsets
  gemm_bf16_wmma<<<dim3((NROWS + BM - 1) / BM, DMODEL / BN), blk, 0, stream>>>(
      qin, ofw, ofb, offb, NROWS, DMODEL, DMODEL, 0);
  // 9) attention-weight logits
  gemm_bf16_wmma<<<dim3((NROWS + BM - 1) / BM, 128 / BN), blk, 0, stream>>>(
      qin, aww, awb_, awlg, NROWS, 128, DMODEL, 0);
  // 10) bilinear sampling
  deform_sample<<<dim3((NROWS * NH) / 8), blk, 0, stream>>>(val, offb, awlg, ref, ca);
  // 11) cproj = ca @ cow^T + cob
  gemm_bf16_wmma<<<dim3((NROWS + BM - 1) / BM, DMODEL / BN), blk, 0, stream>>>(
      ca, cow, cob, cproj, NROWS, DMODEL, DMODEL, 0);
  // 12) q3 = LN(q2 + cproj) [norm1]
  add_layernorm<<<dim3(NROWS / 8), blk, 0, stream>>>(q2, cproj, n1g, n1b, q3, NROWS);
  // 13) ffh = relu(q3 @ l1w^T + l1b)
  gemm_bf16_wmma<<<dim3((NROWS + BM - 1) / BM, DFFN / BN), blk, 0, stream>>>(
      q3, l1w, l1b, ffh, NROWS, DFFN, DMODEL, 1);
  // 14) ffo = ffh @ l2w^T + l2b
  gemm_bf16_wmma<<<dim3((NROWS + BM - 1) / BM, DMODEL / BN), blk, 0, stream>>>(
      ffh, l2w, l2b, ffo, NROWS, DMODEL, DFFN, 0);
  // 15) out = LN(q3 + ffo) [norm3]
  add_layernorm<<<dim3(NROWS / 8), blk, 0, stream>>>(q3, ffo, n3g, n3b, outp, NROWS);

  (void)in_sizes; (void)n_in; (void)out_size; (void)ws_size;
}
